// Encoder_24601572671770
// MI455X (gfx1250) — matrix operand order repaired and checked by reading, not compiled, not run on hardware
//
#include <hip/hip_runtime.h>
#include <hip/hip_bf16.h>

typedef __attribute__((ext_vector_type(16))) _Float16 v16h;
typedef __attribute__((ext_vector_type(8)))  _Float16 v8h;
typedef __attribute__((ext_vector_type(8)))  float    v8f;
typedef int v4i_vs __attribute__((vector_size(16)));

static constexpr int NTOK = 4096;
static constexpr int DIN  = 1024;
static constexpr int DH   = 512;

#define AS1 __attribute__((address_space(1)))
#define AS3 __attribute__((address_space(3)))

#if __has_builtin(__builtin_amdgcn_global_load_async_to_lds_b128)
#define HAVE_ASYNC_LDS 1
#endif

union V16H { v16h v; v8h h[2]; _Float16 e[16]; };

__device__ __forceinline__ v8f wmma_f16(v16h a, v16h b, v8f c) {
  // 8 args: (neg_a, A, neg_b, B, c_mod, C, reuse_a, reuse_b)
  return __builtin_amdgcn_wmma_f32_16x16x32_f16(false, a, false, b, (short)0, c,
                                                false, false);
}

#ifdef HAVE_ASYNC_LDS
__device__ __forceinline__ void wait_asynccnt0() {
#if __has_builtin(__builtin_amdgcn_s_wait_asynccnt)
  __builtin_amdgcn_s_wait_asynccnt(0);
#else
  asm volatile("s_wait_asynccnt 0x0" ::: "memory");
#endif
}
#endif

// ---------------------------------------------------------------------------
// Projection: out = X @ W^T + b   (X:[N,DIN] f32, W:[DH,DIN] f32)
// out is f16; transpose_out=0 -> [N,DH], transpose_out=1 -> [DH,N]
// 8 waves / WG; A tile (16 x DIN) converted to f16 once into LDS, shared.
// ---------------------------------------------------------------------------
__global__ __launch_bounds__(256) void proj_kernel(
    const float* __restrict__ X, const float* __restrict__ W,
    const float* __restrict__ bias, _Float16* __restrict__ out,
    int transpose_out)
{
  __shared__ _Float16 sA[16 * DIN];                 // 32 KB

  const int tid  = threadIdx.x;
  const int lane = tid & 31;
  const int wave = tid >> 5;
  const int l16  = lane & 15;
  const int hi   = lane >> 4;                       // half-wave select
  const int tb   = blockIdx.x;                      // token tile [0,256)
  const int ht   = blockIdx.y * 8 + wave;           // h tile   [0,32)

  {
    const float* src = X + (size_t)tb * 16 * DIN;
    for (int i = tid; i < 16 * DIN; i += 256) sA[i] = (_Float16)src[i];
  }
  __syncthreads();

  v8f acc = {};
  const float* wrow = W + (size_t)(ht * 16 + l16) * DIN;

  for (int ks = 0; ks < DIN; ks += 32) {
    // A fragment (16-bit A 16x32 layout): lane row = l16,
    // halves 0..7 = K ks+hi*8.., halves 8..15 = K ks+16+hi*8..
    V16H a;
    const _Float16* ap = sA + l16 * DIN + ks + hi * 8;
    a.h[0] = *(const v8h*)(ap);
    a.h[1] = *(const v8h*)(ap + 16);
    // B fragment (B 32x16): lane col = l16, K = ks+hi*8.. then ks+16+hi*8..,
    // B[k][n] = W[n][k] -> contiguous along W row.
    V16H b;
    const float* bp = wrow + ks + hi * 8;
#pragma unroll
    for (int j = 0; j < 16; ++j) b.e[j] = (_Float16)bp[j < 8 ? j : j + 8];
    acc = wmma_f16(a.v, b.v, acc);
  }

  const float bv = bias[ht * 16 + l16];
#pragma unroll
  for (int r = 0; r < 8; ++r) acc[r] += bv;

  if (!transpose_out) {
    // C layout: lane col n=l16, VGPR r holds row m = r + hi*8
    _Float16* dst = out + (size_t)(tb * 16 + hi * 8) * DH + ht * 16 + l16;
#pragma unroll
    for (int r = 0; r < 8; ++r) dst[(size_t)r * DH] = (_Float16)acc[r];
  } else {
    union { v8h v; _Float16 e[8]; } pk;
#pragma unroll
    for (int r = 0; r < 8; ++r) pk.e[r] = (_Float16)acc[r];
    *(v8h*)(out + (size_t)(ht * 16 + l16) * NTOK + tb * 16 + hi * 8) = pk.v;
  }
}

// ---------------------------------------------------------------------------
// Flash attention: O = softmax(Q K^T) V.  One WG per 16-query tile.
// 8 waves: per 128-key block, wave w computes S tile w (16x16) -> LDS;
// all waves redo identical row stats; wave w accumulates its 64-col O chunk.
// Q:[N,DH] f16, K:[N,DH] f16, VT:[DH,N] f16, O:[N,DH] f32.
// ---------------------------------------------------------------------------
__global__ __launch_bounds__(256) void attn_kernel(
    const _Float16* __restrict__ Qm, const _Float16* __restrict__ Km,
    const _Float16* __restrict__ VTm, float* __restrict__ Om)
{
  __shared__ _Float16 sQ[16 * DH];                  // 16 KB
  __shared__ float    sS[8][16][16];                // 8 KB raw scores
  __shared__ float    sStat[2][16];                 // corr / l per row

  const int tid  = threadIdx.x;
  const int lane = tid & 31;
  const int wave = tid >> 5;
  const int l16  = lane & 15;
  const int hi   = lane >> 4;
  const int qb   = blockIdx.x;                      // query tile [0,256)
  const int hbase = wave * 64;                      // this wave's O columns

  // ---- stage Q tile into LDS (16 KB pure copy) -----------------------------
#ifdef HAVE_ASYNC_LDS
  {
    const _Float16* src = Qm + (size_t)qb * 16 * DH;
#pragma unroll
    for (int i = 0; i < 4; ++i) {
      const int off = (i * 256 + tid) * 8;          // 8 halves = 16 bytes
      __builtin_amdgcn_global_load_async_to_lds_b128(
          (AS1 v4i_vs*)(src + off), (AS3 v4i_vs*)(sQ + off), 0, 0);
    }
    wait_asynccnt0();
  }
#else
  {
    const uint32_t* s = (const uint32_t*)(Qm + (size_t)qb * 16 * DH);
    uint32_t* d = (uint32_t*)sQ;
    for (int i = tid; i < 16 * DH / 2; i += 256) d[i] = s[i];
  }
#endif
  __syncthreads();

  v8f   o[4] = {{}, {}, {}, {}};
  float m_run = -__builtin_inff();
  float l_run = 0.0f;

  for (int kb = 0; kb < NTOK; kb += 128) {
    // ---- S_w = Qtile @ K[kb+wave*16 .. +16)^T  (K-dim = DH)
    v8f s = {};
    const _Float16* krow = Km + (size_t)(kb + wave * 16 + l16) * DH;
#pragma unroll
    for (int ks = 0; ks < DH; ks += 32) {
      V16H a, b;
      const _Float16* ap = sQ + l16 * DH + ks + hi * 8;
      a.h[0] = *(const v8h*)(ap);
      a.h[1] = *(const v8h*)(ap + 16);
      const _Float16* bp = krow + ks + hi * 8;    // B[d][n]=K[n][d], table order
      b.h[0] = *(const v8h*)(bp);
      b.h[1] = *(const v8h*)(bp + 16);
      s = wmma_f16(a.v, b.v, s);
    }
    if (kb + 128 < NTOK)
      __builtin_prefetch(Km + (size_t)(kb + 128 + wave * 16 + l16) * DH, 0, 1);

#pragma unroll
    for (int r = 0; r < 8; ++r) sS[wave][hi * 8 + r][l16] = s[r];
    __syncthreads();

    // ---- online softmax stats (identical in every wave; lane row = l16)
    float bmax = -__builtin_inff();
#pragma unroll
    for (int f = 0; f < 4; ++f)
#pragma unroll
      for (int j = 0; j < 8; ++j) {
        bmax = fmaxf(bmax, sS[2 * f][l16][hi * 8 + j]);
        bmax = fmaxf(bmax, sS[2 * f + 1][l16][hi * 8 + j]);
      }
    bmax = fmaxf(bmax, __shfl_xor(bmax, 16, 32));
    const float m_new = fmaxf(m_run, bmax);
    const float corr  = __expf(m_run - m_new);

    float psum = 0.0f;
    v16h  pf[4];                                    // P in A-fragment layout
#pragma unroll
    for (int f = 0; f < 4; ++f) {
      V16H p;
#pragma unroll
      for (int j = 0; j < 8; ++j) {
        const float p0 = __expf(sS[2 * f][l16][hi * 8 + j] - m_new);
        const float p1 = __expf(sS[2 * f + 1][l16][hi * 8 + j] - m_new);
        psum += p0 + p1;
        p.e[j]     = (_Float16)p0;                  // kk = f*32 + hi*8 + j
        p.e[j + 8] = (_Float16)p1;                  // kk = f*32 + 16 + hi*8 + j
      }
      pf[f] = p.v;
    }
    psum += __shfl_xor(psum, 16, 32);
    m_run = m_new;
    l_run = l_run * corr + psum;

    if (wave == 0 && hi == 0) sStat[0][l16] = corr;
    __syncthreads();                                // sStat visible; sS reusable

    // ---- rescale O (rows r + hi*8) and accumulate P @ V for 64 columns
#pragma unroll
    for (int r = 0; r < 8; ++r) {
      const float c = sStat[0][hi * 8 + r];
      o[0][r] *= c; o[1][r] *= c; o[2][r] *= c; o[3][r] *= c;
    }
    // f outer / nt inner: same A fragment feeds 4 consecutive WMMAs
#pragma unroll
    for (int f = 0; f < 4; ++f)
#pragma unroll
      for (int nt = 0; nt < 4; ++nt) {
        const _Float16* vrow =
            VTm + (size_t)(hbase + nt * 16 + l16) * NTOK + kb + hi * 8;
        V16H b;                                     // B[kk][n]=V[key][h]=VT[h][key]
        b.h[0] = *(const v8h*)(vrow + f * 32);
        b.h[1] = *(const v8h*)(vrow + f * 32 + 16);
        o[nt] = wmma_f16(pf[f], b.v, o[nt]);
      }
  }

  if (wave == 0 && hi == 0) sStat[1][l16] = l_run;
  __syncthreads();

#pragma unroll
  for (int r = 0; r < 8; ++r) {
    const float inv = 1.0f / sStat[1][hi * 8 + r];
    const size_t row = (size_t)(qb * 16 + hi * 8 + r) * DH + hbase + l16;
    Om[row +  0] = o[0][r] * inv;
    Om[row + 16] = o[1][r] * inv;
    Om[row + 32] = o[2][r] * inv;
    Om[row + 48] = o[3][r] * inv;
  }
}

// ---------------------------------------------------------------------------
extern "C" void kernel_launch(void* const* d_in, const int* in_sizes, int n_in,
                              void* d_out, int out_size, void* d_ws,
                              size_t ws_size, hipStream_t stream) {
  const float* embs = (const float*)d_in[0];
  const float* Wq   = (const float*)d_in[1];
  const float* bq   = (const float*)d_in[2];
  const float* Wk   = (const float*)d_in[3];
  const float* bk   = (const float*)d_in[4];
  const float* Wv   = (const float*)d_in[5];
  const float* bv   = (const float*)d_in[6];
  float* out = (float*)d_out;

  _Float16* q  = (_Float16*)d_ws;                   // [N, DH]
  _Float16* k  = q + (size_t)NTOK * DH;             // [N, DH]
  _Float16* vT = k + (size_t)NTOK * DH;             // [DH, N] (transposed)

  dim3 pgrid(NTOK / 16, DH / (16 * 8), 1);          // 256 x 4
  proj_kernel<<<pgrid, 256, 0, stream>>>(embs, Wq, bq, q, 0);
  proj_kernel<<<pgrid, 256, 0, stream>>>(embs, Wk, bk, k, 0);
  proj_kernel<<<pgrid, 256, 0, stream>>>(embs, Wv, bv, vT, 1);

  attn_kernel<<<NTOK / 16, 256, 0, stream>>>(q, k, vT, out);
}
